// SoftDDT_28518582845631
// MI455X (gfx1250) — matrix operand order repaired and checked by reading, not compiled, not run on hardware
//
#include <hip/hip_runtime.h>
#include <hip/hip_bf16.h>

typedef __bf16 bf16_t;
typedef __bf16 v16bf  __attribute__((ext_vector_type(16)));
typedef __bf16 bf16x8 __attribute__((ext_vector_type(8)));
typedef __bf16 bf16x4 __attribute__((ext_vector_type(4)));
typedef float  v8f    __attribute__((ext_vector_type(8)));
typedef int    i32x4v __attribute__((vector_size(4 * sizeof(int))));

#define BM 128
#define BN 128
#define BK 32
#define LDSB_STRIDE 40   // 32 bf16 + 8 pad = 80 bytes/row -> conflict-free frag reads

#define AS1 __attribute__((address_space(1)))
#define AS3 __attribute__((address_space(3)))

#if defined(__has_builtin)
#  if __has_builtin(__builtin_amdgcn_global_load_async_to_lds_b128)
#    define USE_ASYNC_LDS 1
#  endif
#  if __has_builtin(__builtin_amdgcn_s_wait_asynccnt)
#    define HAVE_WAIT_ASYNC_BUILTIN 1
#  endif
#endif

// 16-byte global -> LDS copy; async (ASYNCcnt) when the toolchain exposes it.
__device__ __forceinline__ void cp16_g2l(bf16_t* l, const bf16_t* g) {
#if defined(USE_ASYNC_LDS)
    __builtin_amdgcn_global_load_async_to_lds_b128(
        (AS1 i32x4v*)g, (AS3 i32x4v*)l, 0, 0);
#else
    *(bf16x8*)l = *(const bf16x8*)g;
#endif
}

// Wait until at most N async-LDS ops remain outstanding (in-order completion).
template <int N>
__device__ __forceinline__ void wait_async_le() {
#if defined(HAVE_WAIT_ASYNC_BUILTIN)
    __builtin_amdgcn_s_wait_asynccnt(N);
#elif defined(USE_ASYNC_LDS)
    if (N == 0) asm volatile("s_wait_asynccnt 0x0" ::: "memory");
    else        asm volatile("s_wait_asynccnt 0x8" ::: "memory");
#endif
}

__device__ __forceinline__ v16bf frag_combine(bf16x8 lo, bf16x8 hi) {
    return __builtin_shufflevector(lo, hi, 0,1,2,3,4,5,6,7,8,9,10,11,12,13,14,15);
}

// ======== one-time f32 -> bf16 hi/lo split (row-padded with zeros) ========
__global__ __launch_bounds__(256) void split_pad(
    const float* __restrict__ src, bf16_t* __restrict__ hi, bf16_t* __restrict__ lo,
    int rows, int cols /*pow2*/, int padrows)
{
    int e4 = (blockIdx.x * 256 + threadIdx.x) * 4;
    if (e4 >= padrows * cols) return;
    int row = e4 / cols;
    float4 v = make_float4(0.f, 0.f, 0.f, 0.f);
    if (row < rows) v = *(const float4*)(src + (size_t)e4);
    float fs[4] = {v.x, v.y, v.z, v.w};
    bf16x4 hv, lv;
#pragma unroll
    for (int i = 0; i < 4; ++i) {
        bf16_t h = (bf16_t)fs[i];
        hv[i] = h;
        lv[i] = (bf16_t)(fs[i] - (float)h);
    }
    *(bf16x4*)(hi + e4) = hv;
    *(bf16x4*)(lo + e4) = lv;
}

// ======== one-time transpose + split: src (K x N f32) -> out[n][k] bf16 hi/lo ========
__global__ __launch_bounds__(256) void transpose_split(
    const float* __restrict__ src, bf16_t* __restrict__ hi, bf16_t* __restrict__ lo,
    int K, int N)
{
    __shared__ float t[32][33];
    const int tx = threadIdx.x & 31, ty = threadIdx.x >> 5;   // 32 x 8
    const int k0 = blockIdx.y * 32, n0 = blockIdx.x * 32;
#pragma unroll
    for (int r = ty; r < 32; r += 8)
        t[r][tx] = src[(size_t)(k0 + r) * N + n0 + tx];
    __syncthreads();
#pragma unroll
    for (int r = ty; r < 32; r += 8) {
        float v = t[tx][r];                       // src[k0+tx][n0+r]
        bf16_t h = (bf16_t)v;
        size_t o = (size_t)(n0 + r) * K + k0 + tx;
        hi[o] = h;
        lo[o] = (bf16_t)(v - (float)h);
    }
}

// ================= bf16x3 split-precision WMMA GEMM (prepacked, double-buffered) ==
// C[m][n] = sum_k A[m][k] * B[n][k]   (A: M x K, B: Npad x K, both row-major bf16 hi/lo)
template <bool SIG>
__global__ __launch_bounds__(256) void softddt_gemm(
    const bf16_t* __restrict__ Ahg, const bf16_t* __restrict__ Alg,
    const bf16_t* __restrict__ Bhg, const bf16_t* __restrict__ Blg,
    const float* __restrict__ bias, float* __restrict__ C,
    int M, int N, int K, int ldc)
{
    __shared__ bf16_t Ah[2][BM][LDSB_STRIDE], Al[2][BM][LDSB_STRIDE];
    __shared__ bf16_t Bh[2][BN][LDSB_STRIDE], Bl[2][BN][LDSB_STRIDE];

    const int tid  = threadIdx.x;
    const int wave = tid >> 5;
    const int lane = tid & 31;
    const int wm   = wave & 3;          // 4 wave-rows * 32 rows
    const int wn   = wave >> 2;         // 2 wave-cols * 64 cols
    const int m0   = blockIdx.y * BM;
    const int n0   = blockIdx.x * BN;

    const int m16     = lane & 15;
    const int hi_half = lane >> 4;
    const int kloA    = hi_half ? 8 : 0;   // A frag: lanes 0-15 K{0..7,16..23}, 16-31 K{8..15,24..31}
    const int kselB   = hi_half ? 8 : 0;   // B frag: lanes 0-15 K{0..7,16..23}, 16-31 K{8..15,24..31}

    // staging geometry: 128 rows x 32 cols bf16 per tile = 512 16B-chunks, 2 per thread
    const int srow0 = tid >> 2,            soff0 = (tid & 3) << 3;
    const int srow1 = (tid + 256) >> 2,    soff1 = ((tid + 256) & 3) << 3;

    // stage one BK-slice (8 async b128 ops per wave) into buffer `buf`
    auto stage = [&](int buf, int k0) {
        cp16_g2l(&Ah[buf][srow0][soff0], Ahg + (size_t)(m0 + srow0) * K + k0 + soff0);
        cp16_g2l(&Ah[buf][srow1][soff1], Ahg + (size_t)(m0 + srow1) * K + k0 + soff1);
        cp16_g2l(&Al[buf][srow0][soff0], Alg + (size_t)(m0 + srow0) * K + k0 + soff0);
        cp16_g2l(&Al[buf][srow1][soff1], Alg + (size_t)(m0 + srow1) * K + k0 + soff1);
        cp16_g2l(&Bh[buf][srow0][soff0], Bhg + (size_t)(n0 + srow0) * K + k0 + soff0);
        cp16_g2l(&Bh[buf][srow1][soff1], Bhg + (size_t)(n0 + srow1) * K + k0 + soff1);
        cp16_g2l(&Bl[buf][srow0][soff0], Blg + (size_t)(n0 + srow0) * K + k0 + soff0);
        cp16_g2l(&Bl[buf][srow1][soff1], Blg + (size_t)(n0 + srow1) * K + k0 + soff1);
    };

    v8f acc[2][4] = {};

    const int nk = K / BK;
    stage(0, 0);                                 // prologue: buffer 0 in flight

    for (int ki = 0; ki < nk; ++ki) {
        const int cur = ki & 1;

        // All waves have finished reading buf[1-cur] (previous iteration's compute);
        // barrier arrival implies their fragment ds_loads completed.
        __syncthreads();
        if (ki + 1 < nk) {
            stage(1 - cur, (ki + 1) * BK);       // overlap next slice with compute
            wait_async_le<8>();                  // oldest 8 (= buf[cur]) retired
        } else {
            wait_async_le<0>();                  // drain: only buf[cur] outstanding
        }
        __syncthreads();                         // buf[cur] visible to all waves

        v16bf aH[2], aL[2], bH[4], bL[4];
#pragma unroll
        for (int mt = 0; mt < 2; ++mt) {
            int r = wm * 32 + mt * 16 + m16;
            aH[mt] = frag_combine(*(const bf16x8*)&Ah[cur][r][kloA],
                                  *(const bf16x8*)&Ah[cur][r][16 + kloA]);
            aL[mt] = frag_combine(*(const bf16x8*)&Al[cur][r][kloA],
                                  *(const bf16x8*)&Al[cur][r][16 + kloA]);
        }
#pragma unroll
        for (int nt = 0; nt < 4; ++nt) {
            int r = wn * 64 + nt * 16 + m16;
            bH[nt] = frag_combine(*(const bf16x8*)&Bh[cur][r][kselB],
                                  *(const bf16x8*)&Bh[cur][r][16 + kselB]);
            bL[nt] = frag_combine(*(const bf16x8*)&Bl[cur][r][kselB],
                                  *(const bf16x8*)&Bl[cur][r][16 + kselB]);
        }

#pragma unroll
        for (int mt = 0; mt < 2; ++mt)
#pragma unroll
            for (int nt = 0; nt < 4; ++nt) {
                acc[mt][nt] = __builtin_amdgcn_wmma_f32_16x16x32_bf16(
                    false, aH[mt], false, bH[nt], (short)0, acc[mt][nt], false, false);
                acc[mt][nt] = __builtin_amdgcn_wmma_f32_16x16x32_bf16(
                    false, aH[mt], false, bL[nt], (short)0, acc[mt][nt], false, false);
                acc[mt][nt] = __builtin_amdgcn_wmma_f32_16x16x32_bf16(
                    false, aL[mt], false, bH[nt], (short)0, acc[mt][nt], false, false);
            }
    }

    // C/D layout: VGPR r -> M = r + 8*lane[4], N = lane&15
#pragma unroll
    for (int mt = 0; mt < 2; ++mt) {
        int rbase = m0 + wm * 32 + mt * 16 + hi_half * 8;
#pragma unroll
        for (int nt = 0; nt < 4; ++nt) {
            int col = n0 + wn * 64 + nt * 16 + m16;
            if (col < N) {
                float bb = SIG ? bias[col] : 0.f;
#pragma unroll
                for (int r = 0; r < 8; ++r) {
                    float v = acc[mt][nt][r];
                    if (SIG) v = 1.0f / (1.0f + __expf(-(v + bb)));
                    C[(size_t)(rbase + r) * ldc + col] = v;
                }
            }
        }
    }
}

// ================= tree pass: leaf probabilities (bf16 hi/lo) + hard routing ======
__global__ __launch_bounds__(256) void softddt_tree(
    const float* __restrict__ V, bf16_t* __restrict__ Ph, bf16_t* __restrict__ Pl,
    int* __restrict__ idxOut)
{
    __shared__ float v[1023];
    const int row = blockIdx.x;
    const float* vr = V + (size_t)row * 1024;
    for (int i = threadIdx.x; i < 1023; i += 256) v[i] = vr[i];
    __syncthreads();

    for (int l = threadIdx.x; l < 1024; l += 256) {
        float p = 1.0f;
#pragma unroll
        for (int d = 0; d < 10; ++d) {
            int j   = l >> (10 - d);
            float g = v[(1 << d) - 1 + j];
            int bit = (l >> (9 - d)) & 1;
            p *= bit ? g : (1.0f - g);
        }
        bf16_t h = (bf16_t)p;
        size_t o = (size_t)row * 1024 + l;
        Ph[o] = h;
        Pl[o] = (bf16_t)(p - (float)h);
    }

    if (threadIdx.x == 0) {
        int i = 0;
#pragma unroll
        for (int d = 0; d < 10; ++d) {
            float g = v[(1 << d) - 1 + i];
            i = 2 * i + (g >= 0.5f ? 1 : 0);
        }
        idxOut[row] = i;
    }
}

// ================= launch =================
extern "C" void kernel_launch(void* const* d_in, const int* in_sizes, int n_in,
                              void* d_out, int out_size, void* d_ws, size_t ws_size,
                              hipStream_t stream)
{
    constexpr int BATCH = 4096, NIN = 2048, NINT = 1023, NLEAF = 1024, NOUT = 1024;

    const float* x    = (const float*)d_in[0];
    const float* W    = (const float*)d_in[1];   // (1023, 2048)
    const float* b    = (const float*)d_in[2];   // (1023,)
    const float* leaf = (const float*)d_in[3];   // (1024, 1024)

    float* out    = (float*)d_out;
    int*   idxOut = (int*)((float*)d_out + (size_t)BATCH * NOUT);

    // workspace layout (bf16 unless noted)
    bf16_t* p = (bf16_t*)d_ws;
    bf16_t* xh  = p;  p += (size_t)BATCH * NIN;      // 16.8 MB
    bf16_t* xl  = p;  p += (size_t)BATCH * NIN;
    bf16_t* Wh  = p;  p += (size_t)1024 * NIN;       // padded to 1024 rows
    bf16_t* Wl  = p;  p += (size_t)1024 * NIN;
    bf16_t* lTh = p;  p += (size_t)NOUT * NLEAF;     // transposed leaf [n][k]
    bf16_t* lTl = p;  p += (size_t)NOUT * NLEAF;
    bf16_t* Ph  = p;  p += (size_t)BATCH * NLEAF;
    bf16_t* Pl  = p;  p += (size_t)BATCH * NLEAF;
    float*  V   = (float*)p;                         // 16 MB f32

    dim3 blk(256);

    // one-time prepack
    split_pad<<<dim3((BATCH * NIN) / 1024), blk, 0, stream>>>(x, xh, xl, BATCH, NIN, BATCH);
    split_pad<<<dim3((1024 * NIN) / 1024), blk, 0, stream>>>(W, Wh, Wl, NINT, NIN, 1024);
    transpose_split<<<dim3(NOUT / 32, NLEAF / 32), blk, 0, stream>>>(leaf, lTh, lTl, NLEAF, NOUT);

    // GEMM1: V = sigmoid(x @ W^T + b)
    softddt_gemm<true><<<dim3(1024 / BN, BATCH / BM), blk, 0, stream>>>(
        xh, xl, Wh, Wl, b, V, BATCH, NINT, NIN, 1024);

    // tree pass: P (bf16 hi/lo) + routing idx
    softddt_tree<<<dim3(BATCH), blk, 0, stream>>>(V, Ph, Pl, idxOut);

    // GEMM2: out = P @ leaf_dist
    softddt_gemm<false><<<dim3(NOUT / BN, BATCH / BM), blk, 0, stream>>>(
        Ph, Pl, lTh, lTl, nullptr, out, BATCH, NOUT, NLEAF, NOUT);
}
